// QuantumAttention_65481071399718
// MI455X (gfx1250) — matrix operand order repaired and checked by reading, not compiled, not run on hardware
//
#include <hip/hip_runtime.h>
#include <hip/hip_bf16.h>

typedef __attribute__((ext_vector_type(16))) _Float16 v16h;
typedef __attribute__((ext_vector_type(8)))  _Float16 v8h;
typedef __attribute__((ext_vector_type(8)))  float    v8f;

#define KB   4        // batch
#define KT   1024     // sequence length
#define KE   512      // embed dim
#define KH   64       // heads
#define KDK  8        // head dim

// Swap data between the two 16-lane halves of a wave32 (xor 16) in the VALU
// (v_permlanex16_b32) instead of the LDS pipe.
__device__ __forceinline__ float xor16f(float v) {
#if __has_builtin(__builtin_amdgcn_permlanex16)
  int iv = __float_as_int(v);
  iv = __builtin_amdgcn_permlanex16(iv, iv, 0x76543210, 0xfedcba98, false, false);
  return __int_as_float(iv);
#else
  return __shfl_xor(v, 16, 32);
#endif
}

// ---------------------------------------------------------------------------
// Kernel 1: convert W_combine (f32 [E][E]) to f16 in workspace.
// ---------------------------------------------------------------------------
__global__ void wcvt_kernel(const float* __restrict__ in,
                            _Float16* __restrict__ out, int n) {
  int i = blockIdx.x * blockDim.x + threadIdx.x;
  if (i < n) out[i] = (_Float16)in[i];
}

// ---------------------------------------------------------------------------
// Kernel 2: fused quantum-projection + flash attention per (b,h).
// proj = cos(x)*cos(theta) in LDS.  Per 32-key block:
//   2x WMMA  S^T = K_tile * Q^T     (K dim = d_k, zero-padded)
//   1x WMMA  O^T = V^T * P^T        (K dim = 32 keys, fully populated)
// Online softmax keeps per-lane (per-query-row) stats; P^T lands in the
// B-operand layout using only xor16 half-swaps (no LDS bounce).
// ---------------------------------------------------------------------------
__global__ __launch_bounds__(256) void qattn_kernel(
    const float* __restrict__ x, const float* __restrict__ theta,
    _Float16* __restrict__ attnH) {
  __shared__ __align__(16) _Float16 projR[KT * KDK];   // [t][d]
  __shared__ __align__(16) _Float16 projT[KDK * KT];   // [d][t]
  __shared__ __align__(16) _Float16 zpad[16];          // zeros for inactive lanes

  const int bh = blockIdx.x;           // 0..255
  const int b  = bh / KH;
  const int h  = bh % KH;
  const float* xb = x + (size_t)b * KT * KE + h * KDK;

  if (threadIdx.x < 16) zpad[threadIdx.x] = (_Float16)0.0f;

  float cth[KDK];
  #pragma unroll
  for (int d = 0; d < KDK; ++d) cth[d] = __cosf(theta[d]);

  for (int i = threadIdx.x; i < KT * KDK; i += 256) {
    int t = i >> 3, d = i & 7;
    float v = __cosf(xb[(size_t)t * KE + d]) * cth[d];
    _Float16 hv = (_Float16)v;
    projR[i] = hv;
    projT[d * KT + t] = hv;
  }
  __syncthreads();

  const int  wv    = threadIdx.x >> 5;
  const int  lane  = threadIdx.x & 31;
  const int  lhalf = lane & 15;
  const bool hi    = lane >= 16;
  const float scale = 0.35355339059327373f;   // 1/sqrt(8)

  // Unconditional fragment load bases: inactive lanes read the zero pad.
  const _Float16* kBase = hi ? zpad : &projR[lane * KDK];
  const int       kStep = hi ? 0 : 16 * KDK;
  const _Float16* vBase = (lhalf < 8) ? &projT[lhalf * KT + (hi ? 16 : 0)] : zpad;
  const int       vStep = (lhalf < 8) ? 32 : 0;   // 32 keys per iteration
  const int       vOff  = (lhalf < 8) ? 8 : 0;    // second 8 keys of this lane's 16

  for (int qi = wv; qi < KT / 16; qi += 8) {
    // B operand of S^T wmma: Q^T, pre-scaled by 1/sqrt(dk).
    v8h q = *(const v8h*)(hi ? zpad : &projR[(qi * 16 + lane) * KDK]);
    v16h bQ;
    #pragma unroll
    for (int k = 0; k < 8; ++k) {
      bQ[k]     = (_Float16)((float)q[k] * scale);
      bQ[8 + k] = (_Float16)0.0f;
    }

    float m_state = -__builtin_inff();
    float l_state = 0.0f;
    v8f o;
    #pragma unroll
    for (int r = 0; r < 8; ++r) o[r] = 0.0f;

    for (int j = 0; j < KT / 32; ++j) {
      // Two key tiles (M=s rows, K=d; 8 valid halves, rest zero).
      v8h kv0 = *(const v8h*)(kBase + (2 * j)     * kStep);
      v8h kv1 = *(const v8h*)(kBase + (2 * j + 1) * kStep);
      v16h aK0, aK1;
      #pragma unroll
      for (int k = 0; k < 8; ++k) {
        aK0[k] = kv0[k]; aK0[8 + k] = (_Float16)0.0f;
        aK1[k] = kv1[k]; aK1[8 + k] = (_Float16)0.0f;
      }
      v8f cz;
      #pragma unroll
      for (int r = 0; r < 8; ++r) cz[r] = 0.0f;
      v8f s0 = __builtin_amdgcn_wmma_f32_16x16x32_f16(
          false, aK0, false, bQ, (short)0, cz, false, false);
      v8f s1 = __builtin_amdgcn_wmma_f32_16x16x32_f16(
          false, aK1, false, bQ, (short)0, cz, false, false);

      // Joint online softmax over the 32-key block.
      float tmax = -__builtin_inff();
      #pragma unroll
      for (int r = 0; r < 8; ++r) tmax = fmaxf(tmax, fmaxf(s0[r], s1[r]));
      tmax = fmaxf(tmax, xor16f(tmax));
      float m_new = fmaxf(m_state, tmax);
      float p0[8], p1[8], psum = 0.0f;
      #pragma unroll
      for (int r = 0; r < 8; ++r) {
        p0[r] = __expf(s0[r] - m_new);
        p1[r] = __expf(s1[r] - m_new);
        psum += p0[r] + p1[r];
      }
      psum += xor16f(psum);
      float alpha = __expf(m_state - m_new);
      l_state = l_state * alpha + psum;
      m_state = m_new;

      // B operand = P^T for 32 keys (K=s32, N=t), fully populated:
      // lanes<16  : halves0-7 = tile0 s0-7 (own),  halves8-15 = tile0 s8-15 (swap)
      // lanes>=16 : halves0-7 = tile1 s0-7 (swap), halves8-15 = tile1 s8-15 (own)
      v16h bP;
      #pragma unroll
      for (int r = 0; r < 8; ++r) {
        float pp0 = xor16f(p0[r]);
        float pp1 = xor16f(p1[r]);
        bP[r]     = (_Float16)(hi ? pp1   : p0[r]);
        bP[8 + r] = (_Float16)(hi ? p1[r] : pp0);
      }

      // A operand = V^T (M=d rows 0-7 valid, K=32 keys), contiguous projT.
      v8h vv0 = *(const v8h*)(vBase + j * vStep);
      v8h vv1 = *(const v8h*)(vBase + j * vStep + vOff);
      v16h aV;
      #pragma unroll
      for (int k = 0; k < 8; ++k) { aV[k] = vv0[k]; aV[8 + k] = vv1[k]; }

      #pragma unroll
      for (int r = 0; r < 8; ++r) o[r] *= alpha;
      // O^T accumulate: lanes0-15, vgpr r = attn[t=lane][d=r]
      o = __builtin_amdgcn_wmma_f32_16x16x32_f16(
          false, aV, false, bP, (short)0, o, false, false);
    }

    if (!hi) {
      float inv = 1.0f / l_state;
      v8h oh;
      #pragma unroll
      for (int r = 0; r < 8; ++r) oh[r] = (_Float16)(o[r] * inv);
      // merged layout [B,T,E], head columns h*8..h*8+7 : one 16B store
      *(v8h*)&attnH[((size_t)(b * KT + qi * 16 + lane)) * KE + h * KDK] = oh;
    }
  }
}

// ---------------------------------------------------------------------------
// Kernel 3: out[4096,512] = attnH(f16) @ W^T + b, f32 accumulate via WMMA.
// 8 waves/block, each owns a 16x32 output tile; K-loop step 32.
// ---------------------------------------------------------------------------
__global__ __launch_bounds__(256) void combine_kernel(
    const _Float16* __restrict__ A, const _Float16* __restrict__ Wh,
    const float* __restrict__ bias, float* __restrict__ out) {
  const int  wv    = threadIdx.x >> 5;
  const int  lane  = threadIdx.x & 31;
  const int  lhalf = lane & 15;
  const bool hi    = lane >= 16;
  const int  m0 = blockIdx.x * 64 + (wv & 3) * 16;
  const int  n0 = blockIdx.y * 64 + (wv >> 2) * 32;

  v8f acc0, acc1;
  #pragma unroll
  for (int r = 0; r < 8; ++r) { acc0[r] = 0.0f; acc1[r] = 0.0f; }

  const _Float16* aBase = A  + (size_t)(m0 + lhalf) * KE + (hi ? 16 : 0);
  const _Float16* wBase = Wh + (size_t)(n0 + lhalf) * KE + (hi ? 16 : 0);

  for (int k0 = 0; k0 < KE; k0 += 32) {
    v8h a0 = *(const v8h*)(aBase + k0);        // K 0-7  / 16-23
    v8h a1 = *(const v8h*)(aBase + k0 + 8);    // K 8-15 / 24-31
    v16h af;
    #pragma unroll
    for (int k = 0; k < 8; ++k) { af[k] = a0[k]; af[8 + k] = a1[k]; }
    v16h bf0 = *(const v16h*)(wBase + k0);
    v16h bf1 = *(const v16h*)(wBase + (size_t)16 * KE + k0);
    acc0 = __builtin_amdgcn_wmma_f32_16x16x32_f16(
        false, af, false, bf0, (short)0, acc0, false, false);
    acc1 = __builtin_amdgcn_wmma_f32_16x16x32_f16(
        false, af, false, bf1, (short)0, acc1, false, false);
  }

  const int orow = m0 + (hi ? 8 : 0);
  const int oc0 = n0 + lhalf;
  const int oc1 = oc0 + 16;
  const float b0 = bias[oc0], b1 = bias[oc1];
  #pragma unroll
  for (int r = 0; r < 8; ++r) {
    out[(size_t)(orow + r) * KE + oc0] = acc0[r] + b0;
    out[(size_t)(orow + r) * KE + oc1] = acc1[r] + b1;
  }
}

// ---------------------------------------------------------------------------
extern "C" void kernel_launch(void* const* d_in, const int* in_sizes, int n_in,
                              void* d_out, int out_size, void* d_ws, size_t ws_size,
                              hipStream_t stream) {
  (void)in_sizes; (void)n_in; (void)out_size; (void)ws_size;
  const float* x     = (const float*)d_in[0];
  const float* theta = (const float*)d_in[1];
  const float* W     = (const float*)d_in[2];
  const float* bias  = (const float*)d_in[3];
  float* out = (float*)d_out;

  _Float16* attnH = (_Float16*)d_ws;                       // 4 MB
  _Float16* Wh    = attnH + (size_t)KB * KT * KE;          // +0.5 MB

  wcvt_kernel<<<(KE * KE + 255) / 256, 256, 0, stream>>>(W, Wh, KE * KE);
  qattn_kernel<<<KB * KH, 256, 0, stream>>>(x, theta, attnH);
  combine_kernel<<<dim3((KB * KT) / 64, KE / 64), 256, 0, stream>>>(attnH, Wh, bias, out);
}
